// SLIM_29111288332534
// MI455X (gfx1250) — compile-verified, matrix operand order checked by reading, not run on hardware
//
#include <hip/hip_runtime.h>

// SLIM predicting_forward: ratings = A[user_ids,:] @ W[:,item_ids]
// M=4096, N=2048, K=8192, fp32 in / fp32 out. Compute-bound (AI ~500 FLOP/B)
// -> bf16 WMMA (v_wmma_f32_16x16x32_bf16) with fp32 accumulation.
// Workspace (96MB): [0,64MB) gathered A bf16 [4096][8192];
//                   [64MB,96MB) gathered+transposed W bf16 [2048][8192].
// Main GEMM uses GLOBAL_LOAD_ASYNC_TO_LDS_B128 (ASYNCcnt) with a 3-stage LDS
// ring so no VGPR staging is needed (fixes round-1 scratch spills).

typedef unsigned short u16;
typedef unsigned int   u32;
typedef __attribute__((ext_vector_type(16))) __bf16 v16bf;
typedef __attribute__((ext_vector_type(8)))  float  v8f;
typedef __attribute__((ext_vector_type(4)))  int    v4i;

typedef v4i __attribute__((address_space(1)))* gv4i_p;   // global
typedef v4i __attribute__((address_space(3)))* lv4i_p;   // LDS

#define N_ITEMS 8192
#define UQ 4096
#define IQ 2048
#define K_DIM 8192

#define BM 128
#define BN 128
#define BK 64
#define LDK 72                 // LDS row stride in halves (144B): bank-conflict-free b128 frag loads
#define NSTAGE (K_DIM / BK)

__device__ __forceinline__ u16 f2bf(float f) {
  u32 u = __float_as_uint(f);
  u32 r = u + 0x7fffu + ((u >> 16) & 1u);   // round-to-nearest-even
  return (u16)(r >> 16);
}
__device__ __forceinline__ u32 pk2(float lo, float hi) {
  return (u32)f2bf(lo) | ((u32)f2bf(hi) << 16);
}

// ---- gfx1250 async global->LDS (16B per lane), tracked by ASYNCcnt ----------
__device__ __forceinline__ void async_load_b128(const u16* gptr, u16* lptr) {
#if __has_builtin(__builtin_amdgcn_global_load_async_to_lds_b128)
  __builtin_amdgcn_global_load_async_to_lds_b128(
      (gv4i_p)gptr, (lv4i_p)lptr, 0, 0);
#else
  u32 l = (u32)(size_t)lptr;   // generic LDS addr low 32 bits == wave LDS offset
  asm volatile("global_load_async_to_lds_b128 %0, %1, off"
               :: "v"(l), "v"(gptr) : "memory");
#endif
}
__device__ __forceinline__ void wait_async_le(int n) {
  if (n == 8) {
#if __has_builtin(__builtin_amdgcn_s_wait_asynccnt)
    __builtin_amdgcn_s_wait_asynccnt(8);
#else
    asm volatile("s_wait_asynccnt 0x8" ::: "memory");
#endif
  } else {
#if __has_builtin(__builtin_amdgcn_s_wait_asynccnt)
    __builtin_amdgcn_s_wait_asynccnt(0);
#else
    asm volatile("s_wait_asynccnt 0x0" ::: "memory");
#endif
  }
}

// ---------------- Prepass 1: gather rows of A, fp32 -> bf16 -------------------
__global__ __launch_bounds__(256) void gather_a_bf16(
    const float* __restrict__ A, const int* __restrict__ uids, u16* __restrict__ Abf) {
  const int u = blockIdx.x;
  const int row = uids[u] & (N_ITEMS - 1);
  const float4* src = (const float4*)(A + (size_t)row * N_ITEMS);
  uint4* dst = (uint4*)(Abf + (size_t)u * K_DIM);
  for (int c = threadIdx.x; c < K_DIM / 8; c += 256) {
    float4 a = src[2 * c + 0];
    float4 b = src[2 * c + 1];
    uint4 o;
    o.x = pk2(a.x, a.y); o.y = pk2(a.z, a.w);
    o.z = pk2(b.x, b.y); o.w = pk2(b.z, b.w);
    dst[c] = o;
  }
}

// ------- Prepass 2: gather columns of W, transpose to Wt[N][K], bf16 ---------
__global__ __launch_bounds__(256) void gather_wt_bf16(
    const float* __restrict__ W, const int* __restrict__ iids, u16* __restrict__ Wt) {
  __shared__ u16 tile[32][34];
  const int n0 = blockIdx.x * 32;
  const int k0 = blockIdx.y * 32;
  const int tx = threadIdx.x;   // n within tile (0..31)
  const int ty = threadIdx.y;   // k chunk (0..7)
  const int col = iids[n0 + tx] & (N_ITEMS - 1);
#pragma unroll
  for (int i = 0; i < 4; ++i) {
    const int k = k0 + ty * 4 + i;
    tile[tx][ty * 4 + i] = f2bf(W[(size_t)k * N_ITEMS + col]);
  }
  __syncthreads();
  const int t = ty * 32 + tx;
  const int rn = t >> 3;          // local n row (0..31)
  const int rk = (t & 7) * 4;     // local k offset (0..28)
  uint2 v;
  v.x = (u32)tile[rn][rk + 0] | ((u32)tile[rn][rk + 1] << 16);
  v.y = (u32)tile[rn][rk + 2] | ((u32)tile[rn][rk + 3] << 16);
  *(uint2*)(Wt + (size_t)(n0 + rn) * K_DIM + k0 + rk) = v;
}

// ------------------------- Main bf16 WMMA GEMM -------------------------------
union FragAB { v16bf v; uint4 q[2]; };

__global__ __launch_bounds__(256, 1) void slim_gemm_wmma(
    const u16* __restrict__ Abf, const u16* __restrict__ Wt, float* __restrict__ C) {
  __shared__ __align__(16) u16 As[3][BM * LDK];
  __shared__ __align__(16) u16 Bs[3][BN * LDK];

  const int tid  = threadIdx.x;
  const int lane = tid & 31;
  const int wid  = tid >> 5;
  const int wm   = wid & 3;    // 4 waves over M -> 32-row slice
  const int wn   = wid >> 2;   // 2 waves over N -> 64-col slice
  const int m0   = blockIdx.y * BM;
  const int n0   = blockIdx.x * BN;

  // global->LDS tiling: 2 threads per row, 4 x b128 (64B) each, async to LDS
  const int lrow = tid >> 1;            // 0..127
  const int lcol = (tid & 1) * 32;      // half-element offset within BK
  const u16* gA = Abf + (size_t)(m0 + lrow) * K_DIM + lcol;
  const u16* gB = Wt  + (size_t)(n0 + lrow) * K_DIM + lcol;

  auto issue = [&](int t) {              // 8 async instrs per wave per stage
    const int s = t % 3;
    const u16* pa = gA + t * BK;
    const u16* pb = gB + t * BK;
    u16* la = &As[s][lrow * LDK + lcol];
    u16* lb = &Bs[s][lrow * LDK + lcol];
#pragma unroll
    for (int i = 0; i < 4; ++i) {
      async_load_b128(pa + i * 8, la + i * 8);
      async_load_b128(pb + i * 8, lb + i * 8);
    }
  };

  v8f acc[8] = {};

  // per-lane fragment addressing (ISA 16-bit 16x32 layout):
  // lanes 0-15: K chunks {0..7, 16..23}; lanes 16-31: {8..15, 24..31}
  const int fhalf = (lane >> 4) << 3;   // 0 or 8 halves
  const int fm    = lane & 15;

  issue(0);
  issue(1);

  for (int t = 0; t < NSTAGE; ++t) {
    const int cur = t % 3;
    // retire stage t (the oldest); stage t+1 may stay in flight
    wait_async_le((t + 1 < NSTAGE) ? 8 : 0);
    __syncthreads();                       // stage t visible to all waves;
                                           // buffer (t+2)%3 is free (read at t-1)
    if (t + 2 < NSTAGE) {
      issue(t + 2);
      if (t + 4 < NSTAGE) {
        __builtin_prefetch(gA + (t + 4) * BK, 0, 1);   // global_prefetch_b8
        __builtin_prefetch(gB + (t + 4) * BK, 0, 1);
      }
    }
#pragma unroll
    for (int kk = 0; kk < BK; kk += 32) {
      FragAB af[2], bfr[4];
#pragma unroll
      for (int i = 0; i < 2; ++i) {
        const u16* base = &As[cur][(wm * 32 + i * 16 + fm) * LDK + kk + fhalf];
        af[i].q[0] = *(const uint4*)base;
        af[i].q[1] = *(const uint4*)(base + 16);
      }
#pragma unroll
      for (int j = 0; j < 4; ++j) {
        const u16* base = &Bs[cur][(wn * 64 + j * 16 + fm) * LDK + kk + fhalf];
        bfr[j].q[0] = *(const uint4*)base;
        bfr[j].q[1] = *(const uint4*)(base + 16);
      }
#pragma unroll
      for (int i = 0; i < 2; ++i)
#pragma unroll
        for (int j = 0; j < 4; ++j)
          acc[i * 4 + j] = __builtin_amdgcn_wmma_f32_16x16x32_bf16(
              false, af[i].v, false, bfr[j].v, (short)0, acc[i * 4 + j], false, false);
    }
  }

  // Epilogue: C/D layout -> VGPR r holds (M = r + 8*(lane>=16), N = lane&15)
  const int mrow = m0 + wm * 32 + ((lane >> 4) << 3);
  const int ncol = n0 + wn * 64 + fm;
#pragma unroll
  for (int i = 0; i < 2; ++i)
#pragma unroll
    for (int j = 0; j < 4; ++j)
#pragma unroll
      for (int r = 0; r < 8; ++r)
        C[(size_t)(mrow + i * 16 + r) * IQ + (ncol + j * 16)] = acc[i * 4 + j][r];
}

// ------------------------------- Launcher ------------------------------------
extern "C" void kernel_launch(void* const* d_in, const int* in_sizes, int n_in,
                              void* d_out, int out_size, void* d_ws, size_t ws_size,
                              hipStream_t stream) {
  const float* A    = (const float*)d_in[0];   // explicit_feedback [8192,8192] f32
  const float* W    = (const float*)d_in[1];   // sparse_weight    [8192,8192] f32
  const int*   uids = (const int*)d_in[2];     // user_ids [4096]
  const int*   iids = (const int*)d_in[3];     // item_ids [2048]
  float* out = (float*)d_out;                  // [4096,2048] f32

  u16* Abf = (u16*)d_ws;                                     // 64 MB
  u16* Wt  = (u16*)((char*)d_ws + (size_t)UQ * K_DIM * 2);   // 32 MB

  gather_a_bf16<<<UQ, 256, 0, stream>>>(A, uids, Abf);

  dim3 gw(IQ / 32, K_DIM / 32);
  dim3 bw(32, 8);
  gather_wt_bf16<<<gw, bw, 0, stream>>>(W, iids, Wt);

  dim3 gg(IQ / BN, UQ / BM);   // (16, 32)
  slim_gemm_wmma<<<gg, 256, 0, stream>>>(Abf, Wt, out);
}
